// TransformerBlock_47863115547351
// MI455X (gfx1250) — matrix operand order repaired and checked by reading, not compiled, not run on hardware
//
#include <hip/hip_runtime.h>
#include <hip/hip_bf16.h>

// ===========================================================================
// Transformer block for MI455X (gfx1250), bf16 WMMA pipeline.
//   Xt = X^T; Q=Xt Wq; K=Xt Wk; RH=(Xt Wv) Wo
//   S = causal(Q K^T); A = softmax(S); X1 = Xt + A RH
//   X2 = X1 + relu(X1 Wmi) Wmo; out = X2^T
// GEMMs: v_wmma_f32_16x16x32_bf16, double-buffered LDS staged with
// global_load_async_to_lds_b128 (ASYNCcnt) when the toolchain provides it.
// Causal structure exploited: upper-triangle score tiles skipped; A@RH
// K-loop limited to k < row-block end (A is exactly 0 beyond).
// ===========================================================================

typedef __bf16 bf16_t;
typedef __attribute__((ext_vector_type(16))) __bf16 v16bf;
typedef __attribute__((ext_vector_type(8)))  float   v8f;
typedef int v4i_ls __attribute__((__vector_size__(16)));   // builtin's pointee type

#define DM   1024
#define NTOK 4096

// -------------------- async global->LDS copy (16 B / lane) -----------------
#if __has_builtin(__builtin_amdgcn_global_load_async_to_lds_b128)
#define HAVE_ASYNC_LDS 1
#else
#define HAVE_ASYNC_LDS 0
#endif

__device__ __forceinline__ void cp16_g2l(bf16_t* l, const bf16_t* g) {
#if HAVE_ASYNC_LDS
  __builtin_amdgcn_global_load_async_to_lds_b128(
      (__attribute__((address_space(1))) v4i_ls*)g,
      (__attribute__((address_space(3))) v4i_ls*)l, 0, 0);
#else
  *(uint4*)l = *(const uint4*)g;
#endif
}

__device__ __forceinline__ void async_wait0() {
#if HAVE_ASYNC_LDS
#if __has_builtin(__builtin_amdgcn_s_wait_asynccnt)
  __builtin_amdgcn_s_wait_asynccnt(0);
#else
  asm volatile("s_wait_asynccnt 0" ::: "memory");
#endif
#endif
}

// ---------------------------------------------------------------------------
// Tiled transpose: src [R,C] f32 -> dst [C,R]; optional f32 and/or bf16 out.
// block (32,8), grid (C/32, R/32). Also used to emit bf16-transposed weights.
// ---------------------------------------------------------------------------
__global__ void k_transpose(const float* __restrict__ src, float* __restrict__ dstf,
                            bf16_t* __restrict__ dstb, int R, int C) {
  __shared__ float tile[32][33];
  int r0 = blockIdx.y * 32, c0 = blockIdx.x * 32;
  int tx = threadIdx.x, ty = threadIdx.y;
  #pragma unroll
  for (int i = 0; i < 32; i += 8)
    tile[ty + i][tx] = src[(size_t)(r0 + ty + i) * C + (c0 + tx)];
  __syncthreads();
  #pragma unroll
  for (int i = 0; i < 32; i += 8) {
    float v = tile[tx][ty + i];
    size_t o = (size_t)(c0 + ty + i) * R + (r0 + tx);
    if (dstf) dstf[o] = v;
    if (dstb) dstb[o] = (bf16_t)v;
  }
}

// ---------------------------------------------------------------------------
// Row softmax: S f32 row (len NTOK, -inf masked) -> bf16 A row (stride astride,
// aliasing the front of each S row; row values held in registers so every read
// of the row completes before any write). One 256-thread block per row.
// ---------------------------------------------------------------------------
__global__ void k_softmax_row(const float* __restrict__ S, bf16_t* __restrict__ A,
                              int astride) {
  __shared__ float red[256];
  int row = blockIdx.x;
  int t = threadIdx.x;
  const float* s = S + (size_t)row * NTOK;
  bf16_t* a = A + (size_t)row * astride;

  float vals[16];
  float mx = -INFINITY;
  #pragma unroll
  for (int c = 0; c < 16; c++) {
    vals[c] = s[(size_t)c * 256 + t];
    mx = fmaxf(mx, vals[c]);
  }
  red[t] = mx; __syncthreads();
  for (int o = 128; o > 0; o >>= 1) {
    if (t < o) red[t] = fmaxf(red[t], red[t + o]);
    __syncthreads();
  }
  mx = red[0];
  __syncthreads();

  float sum = 0.f;
  #pragma unroll
  for (int c = 0; c < 16; c++) {
    vals[c] = __expf(vals[c] - mx);   // masked (-inf) entries -> 0 exactly
    sum += vals[c];
  }
  red[t] = sum; __syncthreads();
  for (int o = 128; o > 0; o >>= 1) {
    if (t < o) red[t] += red[t + o];
    __syncthreads();
  }
  float inv = 1.0f / red[0];
  #pragma unroll
  for (int c = 0; c < 16; c++)
    a[(size_t)c * 256 + t] = (bf16_t)(vals[c] * inv);
}

// ---------------------------------------------------------------------------
// bf16 WMMA GEMM: C(+epilogue) = A[M,K] * Bt^T, with Bt supplied [N,K]
// row-major (so ALL staging is contiguous b128 async copies).
// Block tile 128x128, BK=32, double-buffered LDS; 8 waves, each a 32x64
// sub-tile = 2x4 v_wmma_f32_16x16x32_bf16 accumulators.
// Modes: 0: Cb=bf16(acc) (cbT: store transposed [N,M] ldct)
//        1: Cb=bf16(relu(acc))
//        2: f=Res+acc; Cf=f; Cb=bf16(f)
//        3: Cf=Res+acc
//        4: causal scores: tiles fully above diagonal skip compute, fill -inf;
//           else Cf = (n>m ? -inf : acc)
// causalK: limit K-loop to k < bm0+128 (A rows are 0 beyond the diagonal).
// Dims assumed multiples of 128 (true here).
// ---------------------------------------------------------------------------
#define BM 128
#define BN 128
#define BK 32
#define LDAS (BK + 8)   // 40 elems = 80 B rows (16 B aligned)

__device__ __forceinline__ void stage_tile(
    const bf16_t* __restrict__ A, int lda, const bf16_t* __restrict__ Bt, int ldb,
    int bm0, int bn0, int k0, int tid,
    bf16_t (*As)[LDAS], bf16_t (*Bs)[LDAS]) {
  #pragma unroll
  for (int p = 0; p < 2; p++) {
    int cid = tid + p * 256;
    int r = cid >> 2;           // 0..127
    int c = (cid & 3) * 8;      // 0,8,16,24
    cp16_g2l(&As[r][c], A  + (size_t)(bm0 + r) * lda + k0 + c);
    cp16_g2l(&Bs[r][c], Bt + (size_t)(bn0 + r) * ldb + k0 + c);
  }
}

__global__ __launch_bounds__(256) void k_gemm_bf16(
    const bf16_t* __restrict__ A, int lda,
    const bf16_t* __restrict__ Bt, int ldb,
    int K, int mode, int causalK,
    float* __restrict__ Cf, bf16_t* __restrict__ Cb,
    const float* __restrict__ Res, int ldc, int ldct, int cbT) {
  __shared__ __align__(16) bf16_t AsS[2][BM][LDAS];
  __shared__ __align__(16) bf16_t BsS[2][BN][LDAS];

  int tid  = threadIdx.x;
  int wave = tid >> 5;
  int lane = tid & 31;
  int half = lane >> 4;
  int l16  = lane & 15;

  int bm0 = blockIdx.y * BM;
  int bn0 = blockIdx.x * BN;

  // Fully-masked causal score tile: no GEMM work, just -inf fill.
  if (mode == 4 && bn0 >= bm0 + BM) {
    for (int e = tid; e < BM * BN; e += 256) {
      int m = e >> 7, n = e & 127;
      Cf[(size_t)(bm0 + m) * ldc + (bn0 + n)] = -INFINITY;
    }
    return;
  }

  int kEnd = K;
  if (causalK) { int lim = bm0 + BM; kEnd = lim < K ? lim : K; }
  int nIt = kEnd / BK;

  int wm = (wave >> 1) * 32;   // 0,32,64,96
  int wn = (wave & 1) * 64;    // 0,64

  v8f zero = {};
  v8f acc[2][4];
  #pragma unroll
  for (int i = 0; i < 2; i++)
    #pragma unroll
    for (int j = 0; j < 4; j++) acc[i][j] = zero;

  stage_tile(A, lda, Bt, ldb, bm0, bn0, 0, tid, AsS[0], BsS[0]);

  for (int it = 0; it < nIt; ++it) {
    int buf = it & 1;
    async_wait0();        // my async copies for tile `it` done
    __syncthreads();      // everyone's copies visible; prev compute finished
    if (it + 1 < nIt)
      stage_tile(A, lda, Bt, ldb, bm0, bn0, (it + 1) * BK, tid,
                 AsS[buf ^ 1], BsS[buf ^ 1]);

    bf16_t (*Asb)[LDAS] = AsS[buf];
    bf16_t (*Bsb)[LDAS] = BsS[buf];

    v16bf af[2], bfr[4];
    // A frag (ISA 16-bit A 16x32 layout): lane half h holds K = h*8+0..7 and
    // 16+h*8+0..7 of row l16 -> two contiguous 8-elem LDS runs.
    #pragma unroll
    for (int i = 0; i < 2; i++) {
      const bf16_t* p = &Asb[wm + i * 16 + l16][half * 8];
      #pragma unroll
      for (int e = 0; e < 8; e++) { af[i][e] = p[e]; af[i][8 + e] = p[16 + e]; }
    }
    // B frag: lane = column l16, half h holds K = h*8+0..7 and 16+h*8+0..7.
    #pragma unroll
    for (int j = 0; j < 4; j++) {
      const bf16_t* p = &Bsb[wn + j * 16 + l16][half * 8];
      #pragma unroll
      for (int e = 0; e < 8; e++) { bfr[j][e] = p[e]; bfr[j][8 + e] = p[16 + e]; }
    }
    #pragma unroll
    for (int i = 0; i < 2; i++)
      #pragma unroll
      for (int j = 0; j < 4; j++)
        acc[i][j] = __builtin_amdgcn_wmma_f32_16x16x32_bf16(
            false, af[i], false, bfr[j], (short)0, acc[i][j], false, false);
  }

  // ---- epilogue (C/D layout: VGPR r -> M = half*8 + r, lane%16 -> N) ----
  #pragma unroll
  for (int i = 0; i < 2; i++)
    #pragma unroll
    for (int j = 0; j < 4; j++) {
      int mb = bm0 + wm + i * 16 + half * 8;
      int n  = bn0 + wn + j * 16 + l16;
      #pragma unroll
      for (int r = 0; r < 8; r++) {
        int m = mb + r;
        float v = acc[i][j][r];
        size_t idx = (size_t)m * ldc + n;
        if (mode == 0) {
          if (cbT) Cb[(size_t)n * ldct + m] = (bf16_t)v;
          else     Cb[idx] = (bf16_t)v;
        } else if (mode == 1) {
          Cb[idx] = (bf16_t)fmaxf(v, 0.f);
        } else if (mode == 2) {
          float f = Res[idx] + v; Cf[idx] = f; Cb[idx] = (bf16_t)f;
        } else if (mode == 3) {
          Cf[idx] = Res[idx] + v;
        } else {
          Cf[idx] = (n > m) ? -INFINITY : v;   // mode 4, diagonal-band tiles
        }
      }
    }
}

// ===========================================================================
// Launch
// ===========================================================================
extern "C" void kernel_launch(void* const* d_in, const int* in_sizes, int n_in,
                              void* d_out, int out_size, void* d_ws, size_t ws_size,
                              hipStream_t stream) {
  (void)in_sizes; (void)n_in; (void)out_size; (void)ws_size;
  const float* X = (const float*)d_in[0];
  const float* Wsrc[6] = { (const float*)d_in[1], (const float*)d_in[2],
                           (const float*)d_in[3], (const float*)d_in[4],
                           (const float*)d_in[5], (const float*)d_in[6] };
  float* out = (float*)d_out;

  const size_t MB = 1024ull * 1024ull;
  char* w = (char*)d_ws;
  // Workspace map (peak ~148 MB, with reuse):
  bf16_t* Xt_b = (bf16_t*)(w + 0);          //  8 MB  Xt bf16 [N,d]
  bf16_t* WbT[6];                           // 12 MB  weights bf16, TRANSPOSED [N,K]
  for (int i = 0; i < 6; i++) WbT[i] = (bf16_t*)(w + 8 * MB + (size_t)i * 2 * MB);
  bf16_t* Qb   = (bf16_t*)(w + 20 * MB);    //  8 MB  [NTOK,DM]
  bf16_t* Kb   = (bf16_t*)(w + 28 * MB);    //  8 MB  [NTOK,DM] (B of scores, natural [N,K])
  bf16_t* Vb   = (bf16_t*)(w + 36 * MB);    //  8 MB  Xt*Wv
  bf16_t* RHt  = (bf16_t*)(w + 44 * MB);    //  8 MB  ((Xt*Wv)*Wo)^T  [DM,NTOK]
  float*  Xt_f = (float*)(w + 52 * MB);     // 16 MB  Xt f32 residual
  float*  S    = (float*)(w + 68 * MB);     // 64 MB  causal scores f32
  float*  X1f  = (float*)(w + 132 * MB);    // 16 MB  Xt + A*RH
  bf16_t* Ab   = (bf16_t*)S;                // bf16 A in-place over S rows, stride 2*NTOK
  bf16_t* X1b  = Qb;                        // reuse (Q dead after scores)
  bf16_t* Hb   = Kb;                        // reuse (K dead after scores)
  float*  X2f  = (float*)(w + 36 * MB);     // reuse Vb+RHt (dead by then), 16 MB

  dim3 blk256(256);
  dim3 tblk(32, 8);

  // 1) weights -> bf16, transposed to [N,K] row-major
  for (int i = 0; i < 6; i++)
    k_transpose<<<dim3(DM / 32, DM / 32), tblk, 0, stream>>>(Wsrc[i], nullptr, WbT[i], DM, DM);

  // 2) Xt = X^T (f32 + bf16)
  k_transpose<<<dim3(NTOK / 32, DM / 32), tblk, 0, stream>>>(X, Xt_f, Xt_b, DM, NTOK);

  dim3 gD(DM / BN, NTOK / BM);    // [4096 x 1024] outputs
  dim3 gS(NTOK / BN, NTOK / BM);  // [4096 x 4096] scores

  // 3) Q, K, V' = Xt @ {Wq,Wk,Wv}; RH^T = ((V' @ Wo))^T via transposed store
  k_gemm_bf16<<<gD, blk256, 0, stream>>>(Xt_b, DM, WbT[0], DM, DM, 0, 0,
                                         nullptr, Qb, nullptr, DM, 0, 0);
  k_gemm_bf16<<<gD, blk256, 0, stream>>>(Xt_b, DM, WbT[1], DM, DM, 0, 0,
                                         nullptr, Kb, nullptr, DM, 0, 0);
  k_gemm_bf16<<<gD, blk256, 0, stream>>>(Xt_b, DM, WbT[2], DM, DM, 0, 0,
                                         nullptr, Vb, nullptr, DM, 0, 0);
  k_gemm_bf16<<<gD, blk256, 0, stream>>>(Vb, DM, WbT[3], DM, DM, 0, 0,
                                         nullptr, RHt, nullptr, DM, NTOK, 1);

  // 4) S = causal(Q @ K^T); upper-triangle tiles skip compute entirely
  k_gemm_bf16<<<gS, blk256, 0, stream>>>(Qb, DM, Kb, DM, DM, 4, 0,
                                         S, nullptr, nullptr, NTOK, 0, 0);

  // 5) A = softmax(S), bf16 in place (row stride 2*NTOK bf16 elems)
  k_softmax_row<<<dim3(NTOK), blk256, 0, stream>>>(S, Ab, 2 * NTOK);

  // 6) X1 = Xt + A @ RH; K-loop causally limited (A==0 for k >= bm0+128)
  k_gemm_bf16<<<gD, blk256, 0, stream>>>(Ab, 2 * NTOK, RHt, NTOK, NTOK, 2, 1,
                                         X1f, X1b, Xt_f, DM, 0, 0);

  // 7) H = relu(X1 @ mlp_in)
  k_gemm_bf16<<<gD, blk256, 0, stream>>>(X1b, DM, WbT[4], DM, DM, 1, 0,
                                         nullptr, Hb, nullptr, DM, 0, 0);

  // 8) X2 = X1 + H @ mlp_out
  k_gemm_bf16<<<gD, blk256, 0, stream>>>(Hb, DM, WbT[5], DM, DM, 3, 0,
                                         X2f, nullptr, X1f, DM, 0, 0);

  // 9) out = X2^T
  k_transpose<<<dim3(DM / 32, NTOK / 32), tblk, 0, stream>>>(X2f, out, nullptr, NTOK, DM);
}
